// DCT_CNN_Expert_Core_50053548867795
// MI455X (gfx1250) — compile-verified, matrix operand order checked by reading, not run on hardware
//
#include <hip/hip_runtime.h>
#include <hip/hip_bf16.h>
#include <math.h>

typedef __attribute__((ext_vector_type(16))) _Float16 v16h;
typedef __attribute__((ext_vector_type(8)))  float    v8f;

#define NPATCH_TOTAL 65536      // 16 * 64 * 64
#define PATCH_PER_B  4096
#define NCOLS        1024       // 16 channels * 64 spatial
#define NCHUNK       256        // N columns per workgroup
#define PCHUNK       512        // patches per workgroup
#define NGROUPS      (PCHUNK / 16)
#define MAIN_THREADS 128        // 4 waves of 32

// ---------------- workspace layout (bytes) ----------------
#define WS_MT    0              // f16 [1024][64]  = 131072
#define WS_A2    131072         // f32 [32][1024]  = 131072
#define WS_BIAS  262144         // f32 [1024]      = 4096
#define WS_HSUM  266240         // f32 [16][1024]  = 65536
#define WS_XH    331776         // f16 [65536][64] = 8388608 (fast path only)
#define WS_FAST_BYTES (WS_XH + (size_t)NPATCH_TOTAL * 64 * 2)

// =====================================================================
// Kernel 1: build fused DCT*conv1 matrix M (f16, stored [n][k]),
// effective conv2 spatial-sum weights A2, bias vector, zero Hsum.
// =====================================================================
__global__ void setup_kernel(const float* __restrict__ w1,
                             const float* __restrict__ b1,
                             const float* __restrict__ w2,
                             float* __restrict__ ws_f32,
                             _Float16* __restrict__ Mt)
{
    __shared__ float dct[8][8];
    const int tid = threadIdx.x;
    if (tid < 64) {
        int k = tid >> 3, t = tid & 7;
        float v = (k == 0) ? 0.35355339059327373f
                           : 0.5f * cosf(3.14159265358979323846f * (float)k *
                                         (float)(2 * t + 1) / 16.0f);
        dct[k][t] = v;
    }
    __syncthreads();

    float* A2   = ws_f32 + WS_A2   / 4;
    float* bias = ws_f32 + WS_BIAS / 4;
    float* Hsum = ws_f32 + WS_HSUM / 4;

    for (int e = tid; e < NCOLS * 64; e += blockDim.x) {
        int row = e >> 6, col = e & 63;
        int i = row >> 6, y = (row >> 3) & 7, x = row & 7;
        int t = col >> 3, u = col & 7;
        float acc = 0.0f;
        #pragma unroll
        for (int dy = 0; dy < 3; ++dy) {
            int a = y + dy - 1;
            if (a < 0 || a > 7) continue;
            #pragma unroll
            for (int dx = 0; dx < 3; ++dx) {
                int b = x + dx - 1;
                if (b < 0 || b > 7) continue;
                acc += w1[i * 9 + dy * 3 + dx] * dct[a][t] * dct[b][u];
            }
        }
        Mt[row * 64 + col] = (_Float16)acc;
    }

    for (int e = tid; e < NCOLS; e += blockDim.x)
        bias[e] = b1[e >> 6];

    for (int e = tid; e < 32 * NCOLS; e += blockDim.x) {
        int o = e >> 10, rem = e & 1023;
        int i = rem >> 6, y = (rem >> 3) & 7, x = rem & 7;
        float acc = 0.0f;
        #pragma unroll
        for (int dy = 0; dy < 3; ++dy) {
            int Y = y - dy + 1;
            if (Y < 0 || Y > 7) continue;
            #pragma unroll
            for (int dx = 0; dx < 3; ++dx) {
                int X = x - dx + 1;
                if (X < 0 || X > 7) continue;
                acc += w2[((o * 16 + i) * 3 + dy) * 3 + dx];
            }
        }
        A2[e] = acc;
    }

    for (int e = tid; e < 16 * NCOLS; e += blockDim.x)
        Hsum[e] = 0.0f;
}

// =====================================================================
// Kernel 1b (fast path): stream x (f32 NCHW) -> xh (f16 patch-major)
// Each thread converts one contiguous run of 8 floats (32B read, 16B write).
// =====================================================================
__global__ void convert_kernel(const float* __restrict__ x,
                               _Float16* __restrict__ xh)
{
    const int gid = blockIdx.x * blockDim.x + threadIdx.x;   // 0 .. 524287
    if (gid >= 16 * 512 * 64) return;
    const int px  = gid & 63;
    const int rg  = gid >> 6;          // b*512 + row
    const int b   = rg >> 9;
    const int row = rg & 511;
    const int py  = row >> 3, t = row & 7;

    const float4* src = (const float4*)(x + (size_t)rg * 512 + px * 8);
    float4 lo = src[0], hi = src[1];

    union { _Float16 h[8]; uint4 v; } pk;
    pk.h[0] = (_Float16)lo.x; pk.h[1] = (_Float16)lo.y;
    pk.h[2] = (_Float16)lo.z; pk.h[3] = (_Float16)lo.w;
    pk.h[4] = (_Float16)hi.x; pk.h[5] = (_Float16)hi.y;
    pk.h[6] = (_Float16)hi.z; pk.h[7] = (_Float16)hi.w;

    const int patch = (b << 12) + (py << 6) + px;
    *(uint4*)(xh + (size_t)patch * 64 + t * 8) = pk.v;
}

// =====================================================================
// Kernel 2 (fast path): WMMA GEMM with ASYNC double-buffered LDS staging.
// H1 = relu(P(65536x64) * M^T(64x1024) + b1); Hsum[b][n] += rowsum(tile).
// grid = (4 N-chunks, 8 patch-chunks, 16 batches), 128 threads (4 waves)
// =====================================================================
union Frag { v16h h; unsigned int u[8]; };

__global__ void __launch_bounds__(MAIN_THREADS)
main_gemm_async_kernel(const _Float16* __restrict__ xh,
                       const _Float16* __restrict__ Mt,
                       const float* __restrict__ biasvec,
                       float* __restrict__ Hsum)
{
    __shared__ _Float16 sMt[NCHUNK * 64];   // 32 KB  B slice [n][k]
    __shared__ _Float16 sP[2][16 * 64];     //  4 KB  double-buffered A group
    __shared__ float    sHsum[NCHUNK];
    __shared__ float    sBias[NCHUNK];

    const int tid   = threadIdx.x;
    const int wave  = tid >> 5;
    const int lane  = tid & 31;
    const int row16 = lane & 15;
    const int kh    = lane >> 4;

    const int b          = blockIdx.z;
    const int nBase      = blockIdx.x * NCHUNK;
    const int patchStart = b * PATCH_PER_B + blockIdx.y * PCHUNK; // global patch

    // LDS byte offsets of the two A buffers for this thread's 16B async slot
    const unsigned int lds0 = (unsigned int)(uintptr_t)(&sP[0][0]) + tid * 16;
    const unsigned int lds1 = lds0 + 16 * 64 * 2;

    // ---- issue async copy of patch group 0 immediately (overlaps Mt stage)
    {
        unsigned long long ga =
            (unsigned long long)(uintptr_t)(xh + (size_t)patchStart * 64 + tid * 8);
        asm volatile("global_load_async_to_lds_b128 %0, %1, off"
                     :: "v"(lds0), "v"(ga) : "memory");
    }

    // ---- stage M^T slice (contiguous 32KB), bias, zero accumulators ----
    {
        const unsigned int* src = (const unsigned int*)(Mt + nBase * 64);
        unsigned int* dst = (unsigned int*)sMt;
        for (int idx = tid; idx < NCHUNK * 32; idx += MAIN_THREADS)
            dst[idx] = src[idx];
        for (int idx = tid; idx < NCHUNK; idx += MAIN_THREADS) {
            sBias[idx] = biasvec[nBase + idx];
            sHsum[idx] = 0.0f;
        }
    }
    __syncthreads();

    // ---- persistent B fragments: 4 N-tiles/wave x 2 K-frags ----
    Frag Bf[4][2];
    #pragma unroll
    for (int j = 0; j < 4; ++j) {
        const int n = (wave * 4 + j) * 16 + row16;
        const unsigned int* bsrc = (const unsigned int*)(sMt + n * 64);
        #pragma unroll
        for (int v = 0; v < 8; ++v) {
            const int Kk = (v < 4) ? (kh * 8 + 2 * v) : (16 + kh * 8 + 2 * (v - 4));
            Bf[j][0].u[v] = bsrc[Kk >> 1];
            Bf[j][1].u[v] = bsrc[(Kk + 32) >> 1];
        }
    }

    for (int g = 0; g < NGROUPS; ++g) {
        // own async writes for group g done BEFORE the barrier -> visible to all
        asm volatile("s_wait_asynccnt 0x0" ::: "memory");
        __syncthreads();   // everyone finished group g-1 reads + group g writes

        if (g + 1 < NGROUPS) {   // prefetch group g+1 into the other buffer
            unsigned long long ga = (unsigned long long)(uintptr_t)
                (xh + (size_t)(patchStart + (g + 1) * 16) * 64 + tid * 8);
            const unsigned int dst = ((g + 1) & 1) ? lds1 : lds0;
            asm volatile("global_load_async_to_lds_b128 %0, %1, off"
                         :: "v"(dst), "v"(ga) : "memory");
        }

        // A fragments from buffer g&1
        Frag A0, A1;
        {
            const unsigned int* asrc =
                (const unsigned int*)(&sP[g & 1][0] + row16 * 64);
            #pragma unroll
            for (int v = 0; v < 8; ++v) {
                const int Kk = (v < 4) ? (kh * 8 + 2 * v) : (16 + kh * 8 + 2 * (v - 4));
                A0.u[v] = asrc[Kk >> 1];
                A1.u[v] = asrc[(Kk + 32) >> 1];
            }
        }

        #pragma unroll
        for (int j = 0; j < 4; ++j) {
            v8f c = {};
            c = __builtin_amdgcn_wmma_f32_16x16x32_f16(false, A0.h, false, Bf[j][0].h,
                                                       (short)0, c, false, false);
            c = __builtin_amdgcn_wmma_f32_16x16x32_f16(false, A1.h, false, Bf[j][1].h,
                                                       (short)0, c, false, false);
            const float bias = sBias[wave * 64 + j * 16 + row16];
            float s = 0.0f;
            #pragma unroll
            for (int r = 0; r < 8; ++r) {
                float h = c[r] + bias;
                s += fmaxf(h, 0.0f);
            }
            s += __shfl_xor(s, 16, 32);
            if (lane < 16)
                atomicAdd(&sHsum[(wave * 4 + j) * 16 + lane], s);
        }
    }
    __syncthreads();

    for (int idx = tid; idx < NCHUNK; idx += MAIN_THREADS)
        atomicAdd(&Hsum[b * NCOLS + nBase + idx], sHsum[idx]);
}

// =====================================================================
// Kernel 2 (fallback): in-loop f32->f16 conversion (no xh workspace).
// =====================================================================
__global__ void __launch_bounds__(MAIN_THREADS)
main_gemm_kernel(const float* __restrict__ x,
                 const _Float16* __restrict__ Mt,
                 const float* __restrict__ biasvec,
                 float* __restrict__ Hsum)
{
    __shared__ _Float16 sMt[NCHUNK * 64];
    __shared__ _Float16 sP[16 * 64];
    __shared__ float    sHsum[NCHUNK];
    __shared__ float    sBias[NCHUNK];

    const int tid   = threadIdx.x;
    const int wave  = tid >> 5;
    const int lane  = tid & 31;
    const int row16 = lane & 15;
    const int kh    = lane >> 4;

    const int b         = blockIdx.z;
    const int nBase     = blockIdx.x * NCHUNK;
    const int patchBase = blockIdx.y * PCHUNK;

    {
        const unsigned int* src = (const unsigned int*)(Mt + nBase * 64);
        unsigned int* dst = (unsigned int*)sMt;
        for (int idx = tid; idx < NCHUNK * 32; idx += MAIN_THREADS)
            dst[idx] = src[idx];
        for (int idx = tid; idx < NCHUNK; idx += MAIN_THREADS) {
            sBias[idx] = biasvec[nBase + idx];
            sHsum[idx] = 0.0f;
        }
    }
    __syncthreads();

    Frag Bf[4][2];
    #pragma unroll
    for (int j = 0; j < 4; ++j) {
        const int n = (wave * 4 + j) * 16 + row16;
        const unsigned int* bsrc = (const unsigned int*)(sMt + n * 64);
        #pragma unroll
        for (int v = 0; v < 8; ++v) {
            const int Kk = (v < 4) ? (kh * 8 + 2 * v) : (16 + kh * 8 + 2 * (v - 4));
            Bf[j][0].u[v] = bsrc[Kk >> 1];
            Bf[j][1].u[v] = bsrc[(Kk + 32) >> 1];
        }
    }

    for (int g = 0; g < NGROUPS; ++g) {
        __syncthreads();
        for (int e = tid; e < 16 * 64; e += MAIN_THREADS) {
            const int pl = e >> 6, k = e & 63;
            const int t = k >> 3, u = k & 7;
            const int pb = patchBase + g * 16 + pl;
            const int py = pb >> 6, px = pb & 63;
            const size_t addr = ((size_t)(b * 512 + py * 8 + t)) * 512
                              + (size_t)(px * 8 + u);
            sP[pl * 64 + k] = (_Float16)x[addr];
        }
        __syncthreads();

        Frag A0, A1;
        {
            const unsigned int* asrc = (const unsigned int*)(sP + row16 * 64);
            #pragma unroll
            for (int v = 0; v < 8; ++v) {
                const int Kk = (v < 4) ? (kh * 8 + 2 * v) : (16 + kh * 8 + 2 * (v - 4));
                A0.u[v] = asrc[Kk >> 1];
                A1.u[v] = asrc[(Kk + 32) >> 1];
            }
        }

        #pragma unroll
        for (int j = 0; j < 4; ++j) {
            v8f c = {};
            c = __builtin_amdgcn_wmma_f32_16x16x32_f16(false, A0.h, false, Bf[j][0].h,
                                                       (short)0, c, false, false);
            c = __builtin_amdgcn_wmma_f32_16x16x32_f16(false, A1.h, false, Bf[j][1].h,
                                                       (short)0, c, false, false);
            const float bias = sBias[wave * 64 + j * 16 + row16];
            float s = 0.0f;
            #pragma unroll
            for (int r = 0; r < 8; ++r) {
                float h = c[r] + bias;
                s += fmaxf(h, 0.0f);
            }
            s += __shfl_xor(s, 16, 32);
            if (lane < 16)
                atomicAdd(&sHsum[(wave * 4 + j) * 16 + lane], s);
        }
    }
    __syncthreads();

    for (int idx = tid; idx < NCHUNK; idx += MAIN_THREADS)
        atomicAdd(&Hsum[b * NCOLS + nBase + idx], sHsum[idx]);
}

// =====================================================================
// Kernel 3: out[b,o] = (1/(64*4096)) * dot(A2[o], Hsum[b]) + b2[o]
// =====================================================================
__global__ void epilogue_kernel(const float* __restrict__ A2,
                                const float* __restrict__ Hsum,
                                const float* __restrict__ b2,
                                float* __restrict__ out)
{
    const int t = blockIdx.x * blockDim.x + threadIdx.x;
    if (t >= 16 * 32) return;
    const int b = t >> 5, o = t & 31;
    const float* a = A2   + o * NCOLS;
    const float* h = Hsum + b * NCOLS;
    float acc = 0.0f;
    for (int n = 0; n < NCOLS; ++n)
        acc += a[n] * h[n];
    out[t] = acc * (1.0f / (64.0f * 4096.0f)) + b2[o];
}

// =====================================================================
extern "C" void kernel_launch(void* const* d_in, const int* in_sizes, int n_in,
                              void* d_out, int out_size, void* d_ws, size_t ws_size,
                              hipStream_t stream) {
    const float* x  = (const float*)d_in[0];
    const float* w1 = (const float*)d_in[1];
    const float* b1 = (const float*)d_in[2];
    const float* w2 = (const float*)d_in[3];
    const float* b2 = (const float*)d_in[4];
    float* out = (float*)d_out;

    char* ws = (char*)d_ws;
    _Float16* Mt   = (_Float16*)(ws + WS_MT);
    float*    A2   = (float*)(ws + WS_A2);
    float*    bias = (float*)(ws + WS_BIAS);
    float*    Hsum = (float*)(ws + WS_HSUM);
    _Float16* xh   = (_Float16*)(ws + WS_XH);

    setup_kernel<<<1, 256, 0, stream>>>(w1, b1, w2, (float*)ws, Mt);

    dim3 grid(NCOLS / NCHUNK, PATCH_PER_B / PCHUNK, 16);
    if (ws_size >= WS_FAST_BYTES) {
        convert_kernel<<<(16 * 512 * 64) / 256, 256, 0, stream>>>(x, xh);
        main_gemm_async_kernel<<<grid, MAIN_THREADS, 0, stream>>>(xh, Mt, bias, Hsum);
    } else {
        main_gemm_kernel<<<grid, MAIN_THREADS, 0, stream>>>(x, Mt, bias, Hsum);
    }

    epilogue_kernel<<<1, 512, 0, stream>>>(A2, Hsum, b2, out);
}
